// NormalGRU_16449724745587
// MI455X (gfx1250) — matrix operand order repaired and checked by reading, not compiled, not run on hardware
//
#include <hip/hip_runtime.h>
#include <hip/hip_bf16.h>

// ---------------------------------------------------------------------------
// GRU session model on MI455X (gfx1250, wave32, WMMA).
//
// Shapes: B=1024, T=200, E=100, H=200, V=50000.
//   * bf16 WMMA (v_wmma_f32_16x16x32_bf16), f32 accumulate, f32 state.
//   * GRU: 64 blocks x 13 waves (one wave per 16-wide h-column tile).
//     Per-wave VGPR cap at this occupancy is ~256 (4 waves/SIMD): weights
//     cannot be register-resident. The recurrent matrix Wh (273 KB bf16)
//     lives in CDNA5's 320 KB LDS; x_t and the bf16 h-state are DOUBLE-
//     BUFFERED in LDS (302 KB total) so each timestep needs only ONE
//     barrier, the t+1 gather overlaps compute of t, and A/B fragments are
//     software-pipelined one k-stage ahead (loads get a full stage of
//     latency distance instead of ds_load->wait0->wmma).
//   * Wi (12 fragments) streams from L2 with the same 1-stage pipeline.
//   * Output GEMM: 4 N-tiles per wave per A fragment; HBM-store bound
//     (205 MB @ 23.3 TB/s ~ 9 us floor).
// ---------------------------------------------------------------------------

typedef __attribute__((ext_vector_type(16))) __bf16 v16bf;
typedef __attribute__((ext_vector_type(8)))  __bf16 v8bf;
typedef __attribute__((ext_vector_type(8)))  float  v8f;

#define BATCH   1024
#define SEQ     200
#define EDIM    100
#define HDIM    200
#define VOCAB   50000
#define KE      128   // EDIM padded to multiple of 32 (WMMA K)
#define KH      224   // HDIM padded to multiple of 32
#define CP      208   // per-gate output width padded to multiple of 16
#define NT_H    13    // CP/16 column tiles per gate (== waves per GRU block)
#define NTILES  (VOCAB / 16)   // 3125 output column tiles

#define WH_ELEMS (3 * CP * KH)   // 139,776 bf16 = 279,552 B
#define GRU_LDS_BYTES ((WH_ELEMS + 2 * 16 * KE + 2 * 16 * KH) * 2)  // 302,080 B

// ---- WMMA helpers ---------------------------------------------------------

static __device__ inline v8f wmma_bf16(v16bf a, v16bf b, v8f c) {
    return __builtin_amdgcn_wmma_f32_16x16x32_bf16(
        /*neg_a=*/false, a, /*neg_b=*/false, b,
        /*c_mod=*/(short)0, c, /*reuse_a=*/false, /*reuse_b=*/false);
}

// A fragment (16x32 bf16, row-major source, row stride `stride` elems).
static __device__ inline v16bf load_a_frag(const __bf16* base, int stride,
                                           int lane, int k0) {
    int row = lane & 15;
    int klo = k0 + ((lane >= 16) ? 16 : 0);
    const __bf16* p = base + row * stride + klo;
    v8bf lo = *reinterpret_cast<const v8bf*>(p);
    v8bf hi = *reinterpret_cast<const v8bf*>(p + 8);
    v16bf r;
#pragma unroll
    for (int i = 0; i < 8; ++i) { r[i] = lo[i]; r[i + 8] = hi[i]; }
    return r;
}

// B fragment (32x16 bf16) from a [columns][K] K-contiguous layout.
static __device__ inline v16bf load_b_frag(const __bf16* base, int kstride,
                                           int lane, int c0, int k0) {
    int c = c0 + (lane & 15);
    int k = k0 + ((lane >= 16) ? 16 : 0);
    return *reinterpret_cast<const v16bf*>(base + c * kstride + k);
}

// ---- one-time weight repack: f32 -> padded bf16, per-gate, K-contiguous ---

__global__ void conv_wi_kernel(const float* __restrict__ Wi, __bf16* __restrict__ WiT) {
    int n = 3 * CP * KE;
    for (int i = blockIdx.x * blockDim.x + threadIdx.x; i < n;
         i += gridDim.x * blockDim.x) {
        int g = i / (CP * KE);
        int rem = i % (CP * KE);
        int c = rem / KE, k = rem % KE;
        float v = (c < HDIM && k < EDIM) ? Wi[k * (3 * HDIM) + g * HDIM + c] : 0.0f;
        WiT[i] = (__bf16)v;
    }
}

__global__ void conv_wh_kernel(const float* __restrict__ Wh, __bf16* __restrict__ WhT) {
    int n = WH_ELEMS;
    for (int i = blockIdx.x * blockDim.x + threadIdx.x; i < n;
         i += gridDim.x * blockDim.x) {
        int g = i / (CP * KH);
        int rem = i % (CP * KH);
        int c = rem / KH, k = rem % KH;
        float v = (c < HDIM && k < HDIM) ? Wh[k * (3 * HDIM) + g * HDIM + c] : 0.0f;
        WhT[i] = (__bf16)v;
    }
}

__global__ void conv_wo_kernel(const float* __restrict__ Wo, __bf16* __restrict__ WoT) {
    long long n = (long long)VOCAB * KH;
    for (long long i = blockIdx.x * (long long)blockDim.x + threadIdx.x; i < n;
         i += (long long)gridDim.x * blockDim.x) {
        int c = (int)(i / KH), k = (int)(i % KH);
        float v = (k < HDIM) ? Wo[(long long)k * VOCAB + c] : 0.0f;
        WoT[i] = (__bf16)v;
    }
}

// ---- recurrent kernel -----------------------------------------------------

__launch_bounds__(NT_H * 32)
__global__ void gru_kernel(const int* __restrict__ q,
                           const float* __restrict__ embed,
                           const __bf16* __restrict__ WiT,
                           const __bf16* __restrict__ WhT,
                           const float* __restrict__ bi,
                           const float* __restrict__ bhn,
                           __bf16* __restrict__ hOut) {
    extern __shared__ __bf16 smem[];
    __bf16* whl = smem;                        // [3][CP][KH]  273 KB
    __bf16* xb0 = smem + WH_ELEMS;             // x ping       4 KB
    __bf16* xb1 = xb0 + 16 * KE;               // x pong       4 KB
    __bf16* hb0 = xb1 + 16 * KE;               // h ping       7 KB
    __bf16* hb1 = hb0 + 16 * KH;               // h pong       7 KB

    const int tid  = threadIdx.x;
    const int wid  = tid >> 5;          // 0..12 -> owned column tile
    const int lane = tid & 31;
    const int b0   = blockIdx.x * 16;

    // stage Wh into LDS (vectorized 16B copies); zero both h buffers
    {
        const uint4* src = (const uint4*)WhT;
        uint4*       dst = (uint4*)whl;
        for (int i = tid; i < WH_ELEMS / 8; i += NT_H * 32) dst[i] = src[i];
    }
    for (int i = tid; i < 16 * KH; i += NT_H * 32) {
        hb0[i] = (__bf16)0.0f;
        hb1[i] = (__bf16)0.0f;
    }
    // prologue gather: x for t=0 into xb0
    if (tid < 256) {
        int flat = tid * 8;
        int row  = flat >> 7;
        int cb   = flat & 127;
        int id   = q[(b0 + row) * SEQ + 0];
        const float* erow = embed + (long long)id * EDIM;
#pragma unroll
        for (int e = 0; e < 8; ++e) {
            int c = cb + e;
            float v = (id != 0 && c < EDIM) ? erow[c] : 0.0f;
            xb0[row * KE + c] = (__bf16)v;
        }
    }

    const int c0  = wid * 16;
    const int col = c0 + (lane & 15);
    const float b_r  = (col < HDIM) ? bi[col]            : 0.0f;
    const float b_z  = (col < HDIM) ? bi[HDIM + col]     : 0.0f;
    const float b_n  = (col < HDIM) ? bi[2 * HDIM + col] : 0.0f;
    const float b_hn = (col < HDIM) ? bhn[col]           : 0.0f;

    const __bf16* wir = WiT + 0 * CP * KE;
    const __bf16* wiz = WiT + 1 * CP * KE;
    const __bf16* win = WiT + 2 * CP * KE;
    const __bf16* whr = whl + 0 * CP * KH;
    const __bf16* whz = whl + 1 * CP * KH;
    const __bf16* whn = whl + 2 * CP * KH;

    v8f h = {};                         // persistent f32 state, D layout
    __syncthreads();

    for (int t = 0; t < SEQ; ++t) {
        const __bf16* xcur  = (t & 1) ? xb1 : xb0;
        __bf16*       xnext = (t & 1) ? xb0 : xb1;
        const __bf16* hprev = (t & 1) ? hb0 : hb1;
        __bf16*       hcur  = (t & 1) ? hb1 : hb0;

        // ---- gather x for step t+1 (overlaps compute of step t) -----------
        if (t + 1 < SEQ && tid < 256) {
            int flat = tid * 8;
            int row  = flat >> 7;
            int cb   = flat & 127;
            int id   = q[(b0 + row) * SEQ + (t + 1)];
            const float* erow = embed + (long long)id * EDIM;
#pragma unroll
            for (int e = 0; e < 8; ++e) {
                int c = cb + e;
                float v = (id != 0 && c < EDIM) ? erow[c] : 0.0f;
                xnext[row * KE + c] = (__bf16)v;
            }
        }

        v8f gir = {}, giz = {}, gin = {}, ghr = {}, ghz = {}, ghn = {};

        // ---- gi: 1-stage software pipeline over KE (Wi from L2) -----------
        {
            v16bf a  = load_a_frag(xcur, KE, lane, 0);
            v16bf br = load_b_frag(wir, KE, lane, c0, 0);
            v16bf bz = load_b_frag(wiz, KE, lane, c0, 0);
            v16bf bn = load_b_frag(win, KE, lane, c0, 0);
#pragma unroll
            for (int k = 0; k < KE / 32 - 1; ++k) {
                v16bf a2  = load_a_frag(xcur, KE, lane, (k + 1) * 32);
                v16bf br2 = load_b_frag(wir, KE, lane, c0, (k + 1) * 32);
                v16bf bz2 = load_b_frag(wiz, KE, lane, c0, (k + 1) * 32);
                v16bf bn2 = load_b_frag(win, KE, lane, c0, (k + 1) * 32);
                gir = wmma_bf16(a, br, gir);
                giz = wmma_bf16(a, bz, giz);
                gin = wmma_bf16(a, bn, gin);
                a = a2; br = br2; bz = bz2; bn = bn2;
            }
            gir = wmma_bf16(a, br, gir);
            giz = wmma_bf16(a, bz, giz);
            gin = wmma_bf16(a, bn, gin);
        }
        // ---- gh: 1-stage software pipeline over KH (Wh from LDS) ----------
        {
            v16bf a  = load_a_frag(hprev, KH, lane, 0);
            v16bf br = load_b_frag(whr, KH, lane, c0, 0);
            v16bf bz = load_b_frag(whz, KH, lane, c0, 0);
            v16bf bn = load_b_frag(whn, KH, lane, c0, 0);
#pragma unroll
            for (int k = 0; k < KH / 32 - 1; ++k) {
                v16bf a2  = load_a_frag(hprev, KH, lane, (k + 1) * 32);
                v16bf br2 = load_b_frag(whr, KH, lane, c0, (k + 1) * 32);
                v16bf bz2 = load_b_frag(whz, KH, lane, c0, (k + 1) * 32);
                v16bf bn2 = load_b_frag(whn, KH, lane, c0, (k + 1) * 32);
                ghr = wmma_bf16(a, br, ghr);
                ghz = wmma_bf16(a, bz, ghz);
                ghn = wmma_bf16(a, bn, ghn);
                a = a2; br = br2; bz = bz2; bn = bn2;
            }
            ghr = wmma_bf16(a, br, ghr);
            ghz = wmma_bf16(a, bz, ghz);
            ghn = wmma_bf16(a, bn, ghn);
        }

        // ---- fused GRU update (D-layout aligned, pure registers) ----------
        v8f hnew;
#pragma unroll
        for (int r = 0; r < 8; ++r) {
            float rr = 1.0f / (1.0f + __expf(-(gir[r] + b_r + ghr[r])));
            float zz = 1.0f / (1.0f + __expf(-(giz[r] + b_z + ghz[r])));
            float nn = tanhf(gin[r] + b_n + rr * (ghn[r] + b_hn));
            hnew[r] = (1.0f - zz) * nn + zz * h[r];
        }
        h = hnew;

        // publish bf16 state to the ping-pong buffer
#pragma unroll
        for (int r = 0; r < 8; ++r) {
            int row = r + ((lane >= 16) ? 8 : 0);
            hcur[row * KH + col] = (__bf16)hnew[r];
        }

        __syncthreads();   // single barrier per timestep
    }

    const __bf16* hlast = ((SEQ - 1) & 1) ? hb1 : hb0;
    for (int i = tid; i < 16 * KH; i += NT_H * 32) {
        int row = i / KH, cc = i % KH;
        hOut[(long long)(b0 + row) * KH + cc] = hlast[i];
    }
}

// ---- output projection: logits = h_last @ Wo + bo -------------------------

__launch_bounds__(256)
__global__ void out_gemm_kernel(const __bf16* __restrict__ hOut,
                                const __bf16* __restrict__ WoT,
                                const float* __restrict__ bo,
                                float* __restrict__ out) {
    const int lane   = threadIdx.x & 31;
    const int wid    = threadIdx.x >> 5;
    const int ntile0 = (blockIdx.x * 8 + wid) * 4;
    const int m0     = blockIdx.y * 16;
    if (ntile0 >= NTILES) return;              // wave-uniform

    const __bf16* A = hOut + (long long)m0 * KH;
    v8f acc[4] = {};
#pragma unroll
    for (int k = 0; k < KH / 32; ++k) {
        v16bf a = load_a_frag(A, KH, lane, k * 32);
#pragma unroll
        for (int n = 0; n < 4; ++n) {
            if (ntile0 + n < NTILES) {         // wave-uniform
                v16bf b = load_b_frag(WoT, KH, lane, (ntile0 + n) * 16, k * 32);
                acc[n] = wmma_bf16(a, b, acc[n]);
            }
        }
    }
#pragma unroll
    for (int n = 0; n < 4; ++n) {
        if (ntile0 + n < NTILES) {
            int cc = (ntile0 + n) * 16 + (lane & 15);
            float bias = bo[cc];
#pragma unroll
            for (int r = 0; r < 8; ++r) {
                int row = m0 + r + ((lane >= 16) ? 8 : 0);
                out[(long long)row * VOCAB + cc] = acc[n][r] + bias;
            }
        }
    }
}

// ---- launch ---------------------------------------------------------------

extern "C" void kernel_launch(void* const* d_in, const int* in_sizes, int n_in,
                              void* d_out, int out_size, void* d_ws, size_t ws_size,
                              hipStream_t stream) {
    const int*   q     = (const int*)  d_in[0];
    const float* embed = (const float*)d_in[1];
    const float* Wi    = (const float*)d_in[2];
    const float* bi    = (const float*)d_in[3];
    const float* Wh    = (const float*)d_in[4];
    const float* bhn   = (const float*)d_in[5];
    const float* Wo    = (const float*)d_in[6];
    const float* bo    = (const float*)d_in[7];
    float* out = (float*)d_out;

    char* ws = (char*)d_ws;
    const size_t wiT_bytes = (size_t)3 * CP * KE * 2;            // 159,744
    const size_t whT_bytes = (size_t)WH_ELEMS * 2;               // 279,552
    const size_t woT_bytes = (size_t)VOCAB * KH * 2;             // 22.4 MB
    __bf16* WiT = (__bf16*)(ws);
    __bf16* WhT = (__bf16*)(ws + wiT_bytes);
    __bf16* WoT = (__bf16*)(ws + wiT_bytes + whT_bytes);
    __bf16* hO  = (__bf16*)(ws + wiT_bytes + whT_bytes + woT_bytes);

    conv_wi_kernel<<<(3 * CP * KE + 255) / 256, 256, 0, stream>>>(Wi, WiT);
    conv_wh_kernel<<<(WH_ELEMS + 255) / 256, 256, 0, stream>>>(Wh, WhT);
    conv_wo_kernel<<<4096, 256, 0, stream>>>(Wo, WoT);

    // allow the 302 KB dynamic-LDS allocation (CDNA5: 320 KB per workgroup)
    (void)hipFuncSetAttribute((const void*)gru_kernel,
                              hipFuncAttributeMaxDynamicSharedMemorySize,
                              GRU_LDS_BYTES);

    gru_kernel<<<BATCH / 16, NT_H * 32, GRU_LDS_BYTES, stream>>>(
        q, embed, WiT, WhT, bi, bhn, hO);

    dim3 grid((NTILES + 31) / 32, BATCH / 16);
    out_gemm_kernel<<<grid, 256, 0, stream>>>(hO, WoT, bo, out);
}
